// UpCSPN_16200616640827
// MI455X (gfx1250) — matrix operand order repaired and checked by reading, not compiled, not run on hardware
//
#include <hip/hip_runtime.h>
#include <hip/hip_bf16.h>

// ---------------------------------------------------------------------------
// UpCSPN for MI455X (gfx1250): implicit-GEMM bf16 WMMA convs + streaming tail.
// conv1 (5x5, K=800) and conv2/gen/conv3 (fused 3x3, N=48 padded) run on
// v_wmma_f32_16x16x32_bf16. Interior waves take a fast path where every
// A-fragment load is a global_load_b128 with an immediate offset (no per-tap
// predication / address math); border waves take the predicated path.
// Weights are BN-folded + pre-swizzled to the wave32 B-matrix lane layout and
// staged in LDS. Intermediates (~168 MB) fit the 192 MB L2; the 256 MB output
// is written with non-temporal stores so it doesn't evict them.
// ---------------------------------------------------------------------------

typedef __attribute__((ext_vector_type(16))) __bf16 v16bf;
typedef __attribute__((ext_vector_type(8)))  float  v8f;

#define BB   8
#define CC   32
#define HH   256
#define WW   256
#define HWSZ (HH * WW)
#define EPSB 1e-5f

__device__ __forceinline__ __bf16 f2bf(float f) {
    union { float f; unsigned u; } a; a.f = f;
    unsigned r = a.u + 0x7FFFu + ((a.u >> 16) & 1u);   // round-to-nearest-even
    unsigned short h = (unsigned short)(r >> 16);
    return __builtin_bit_cast(__bf16, h);
}

// ---------------------------------------------------------------------------
// K0: s = x + guide + depth, NCHW fp32 -> NHWC bf16, LDS-tiled transpose.
// Block = 256 threads handles 32 pixels x 32 channels. Reads are planar-
// coalesced (consecutive pixels per lane), writes are contiguous 64B NHWC.
// ---------------------------------------------------------------------------
__global__ void __launch_bounds__(256)
k_fuse(const float* __restrict__ x, const float* __restrict__ g,
       const float* __restrict__ d, __bf16* __restrict__ s16) {
    __shared__ float tile[32][33];
    const int t  = threadIdx.x;
    const int b  = blockIdx.y;
    const int p0 = blockIdx.x * 32;

#pragma unroll
    for (int k = 0; k < 4; ++k) {
        int c  = (t >> 5) + k * 8;
        int pp = t & 31;
        size_t gi = ((size_t)(b * CC + c)) * HWSZ + p0 + pp;
        tile[c][pp] = x[gi] + g[gi] + d[gi];
    }
    __syncthreads();
#pragma unroll
    for (int k = 0; k < 4; ++k) {
        int pp = (t >> 5) + k * 8;
        int c  = t & 31;
        s16[((size_t)(b * HWSZ + p0 + pp)) * CC + c] = f2bf(tile[c][pp]);
    }
}

// ---------------------------------------------------------------------------
// Pack conv1 weights (32x32x5x5, OIHW) with BN1 scale folded into the wave32
// B-matrix lane layout: idx = ((t*2 + n/16)*32 + (n%16) + 16*((c/8)%2))*16 + 8*(c/16) + c%8
// ---------------------------------------------------------------------------
__global__ void k_pack1(const float* __restrict__ w, const float* __restrict__ gg,
                        const float* __restrict__ bb, const float* __restrict__ mm,
                        const float* __restrict__ vv,
                        __bf16* __restrict__ wout, float* __restrict__ bias) {
    int i = blockIdx.x * blockDim.x + threadIdx.x;
    if (i >= 25 * 32 * 32) return;
    int c = i % 32, n = (i / 32) % 32, t = i / (32 * 32);
    float s = gg[n] * rsqrtf(vv[n] + EPSB);
    float val = w[(n * 32 + c) * 25 + t] * s;
    int idx = ((t * 2 + (n >> 4)) * 32 + (n & 15) + (((c >> 3) & 1) << 4)) * 16 + ((c >> 4) << 3) + (c & 7);
    wout[idx] = f2bf(val);
    if (t == 0 && c == 0) bias[n] = bb[n] - mm[n] * s;
}

// ---------------------------------------------------------------------------
// Pack fused 3x3 weights: n=0 -> conv2 (1x1 embedded at center tap, BN2),
// n=1..8 -> gen_w (BNg), n=9..39 -> conv3_w (BN3), n=40..47 -> zero padding.
// ---------------------------------------------------------------------------
__global__ void k_pack2(const float* __restrict__ w2, const float* __restrict__ g2,
                        const float* __restrict__ b2, const float* __restrict__ m2,
                        const float* __restrict__ v2,
                        const float* __restrict__ wg, const float* __restrict__ gg,
                        const float* __restrict__ bg, const float* __restrict__ mg,
                        const float* __restrict__ vg,
                        const float* __restrict__ w3, const float* __restrict__ g3,
                        const float* __restrict__ b3, const float* __restrict__ m3,
                        const float* __restrict__ v3,
                        __bf16* __restrict__ wout, float* __restrict__ bias) {
    int i = blockIdx.x * blockDim.x + threadIdx.x;
    if (i >= 9 * 48 * 32) return;
    int c = i % 32, n = (i / 32) % 48, t = i / (32 * 48);
    float val = 0.f, bi = 0.f;
    if (n == 0) {
        float s = g2[0] * rsqrtf(v2[0] + EPSB);
        bi  = b2[0] - m2[0] * s;
        val = (t == 4) ? w2[c] * s : 0.f;          // 1x1 conv = center tap only
    } else if (n <= 8) {
        int o = n - 1;
        float s = gg[o] * rsqrtf(vg[o] + EPSB);
        bi  = bg[o] - mg[o] * s;
        val = wg[(o * 32 + c) * 9 + t] * s;
    } else if (n <= 39) {
        int o = n - 9;
        float s = g3[o] * rsqrtf(v3[o] + EPSB);
        bi  = b3[o] - m3[o] * s;
        val = w3[(o * 32 + c) * 9 + t] * s;
    }
    int idx = ((t * 3 + (n >> 4)) * 32 + (n & 15) + (((c >> 3) & 1) << 4)) * 16 + ((c >> 4) << 3) + (c & 7);
    wout[idx] = f2bf(val);
    if (t == 0 && c == 0) bias[n] = bi;
}

// ---------------------------------------------------------------------------
// conv1: 5x5, 32->32, implicit GEMM. Wave = 16 pixels (M) x 32 channels (N,
// 2 tiles). K loop = 25 taps x 32 channels -> 50 WMMAs/wave.
// ---------------------------------------------------------------------------
__global__ void __launch_bounds__(128)
k_conv1(const __bf16* __restrict__ s16, const __bf16* __restrict__ w1,
        const float* __restrict__ bias1, __bf16* __restrict__ x1) {
    __shared__ unsigned wlds[12800];                     // 25*2*32*16 bf16
    const unsigned* wsrc = (const unsigned*)w1;
    for (int i = threadIdx.x; i < 12800; i += 128) wlds[i] = wsrc[i];
    __syncthreads();

    const int wave = threadIdx.x >> 5;
    const int lane = threadIdx.x & 31;
    const int half = lane >> 4;
    const int m    = lane & 15;
    const int b  = blockIdx.z;
    const int y  = blockIdx.y;
    const int xw = blockIdx.x * 64 + wave * 16;

    v8f acc0 = {}; v8f acc1 = {};
    const uint4* wq  = (const uint4*)wlds;
    // per-lane pixel base, uint4 (16B) units; one pixel = 4 uint4
    const uint4* ap0 = (const uint4*)(s16 + ((size_t)((b * HH + y) * WW + xw + m)) * CC);

    const bool interior = (y >= 2) && (y <= HH - 3) && (xw >= 2) && (xw <= WW - 18);
    if (interior) {
        // Fast path: all 25 taps in-bounds -> immediate-offset b128 loads.
#pragma unroll
        for (int t = 0; t < 25; ++t) {
            const int dy = t / 5 - 2, dx = t % 5 - 2;
            const int off = (dy * WW + dx) * 4;          // uint4 units
            union { v16bf v; uint4 q[2]; } af;
            af.q[0] = ap0[off + half];
            af.q[1] = ap0[off + 2 + half];
            union { v16bf v; uint4 q[2]; } bf0, bf1;
            const int i0 = ((t * 2 + 0) * 32 + lane) * 2;
            const int i1 = ((t * 2 + 1) * 32 + lane) * 2;
            bf0.q[0] = wq[i0]; bf0.q[1] = wq[i0 + 1];
            bf1.q[0] = wq[i1]; bf1.q[1] = wq[i1 + 1];
            acc0 = __builtin_amdgcn_wmma_f32_16x16x32_bf16(false, af.v, false, bf0.v,
                                                           (short)0, acc0, false, false);
            acc1 = __builtin_amdgcn_wmma_f32_16x16x32_bf16(false, af.v, false, bf1.v,
                                                           (short)0, acc1, false, false);
        }
    } else {
        for (int t = 0; t < 25; ++t) {
            int dy = t / 5 - 2, dx = t % 5 - 2;
            int py = y + dy, px = xw + m + dx;
            union { v16bf v; uint4 q[2]; } af;
            if (py >= 0 && py < HH && px >= 0 && px < WW) {
                const uint4* ap = (const uint4*)(s16 + ((size_t)((b * HH + py) * WW + px)) * CC);
                af.q[0] = ap[half];
                af.q[1] = ap[2 + half];
            } else {
                af.q[0] = make_uint4(0, 0, 0, 0);
                af.q[1] = make_uint4(0, 0, 0, 0);
            }
            union { v16bf v; uint4 q[2]; } bf0, bf1;
            int i0 = ((t * 2 + 0) * 32 + lane) * 2;
            int i1 = ((t * 2 + 1) * 32 + lane) * 2;
            bf0.q[0] = wq[i0]; bf0.q[1] = wq[i0 + 1];
            bf1.q[0] = wq[i1]; bf1.q[1] = wq[i1 + 1];
            acc0 = __builtin_amdgcn_wmma_f32_16x16x32_bf16(false, af.v, false, bf0.v,
                                                           (short)0, acc0, false, false);
            acc1 = __builtin_amdgcn_wmma_f32_16x16x32_bf16(false, af.v, false, bf1.v,
                                                           (short)0, acc1, false, false);
        }
    }
    // D layout: lane holds N = m (+16), VGPR r holds M = r + 8*half
    const float b0 = bias1[m], b1 = bias1[m + 16];
#pragma unroll
    for (int r = 0; r < 8; ++r) {
        int mm = r + 8 * half;
        size_t pb = ((size_t)((b * HH + y) * WW + xw + mm)) * CC;
        float v0 = acc0[r] + b0;
        float v1 = acc1[r] + b1;
        if (v0 < 0.f) v0 = 0.f;
        if (v1 < 0.f) v1 = 0.f;
        x1[pb + m]      = f2bf(v0);
        x1[pb + m + 16] = f2bf(v1);
    }
}

// ---------------------------------------------------------------------------
// Fused conv2/gen/conv3: 3x3, 32->48(padded), 27 WMMAs/wave, routed outputs.
// ---------------------------------------------------------------------------
__device__ __forceinline__ void route(int n, int b, int sp, float v,
                                      float* __restrict__ dep, float* __restrict__ graw,
                                      float* __restrict__ mid) {
    size_t gpix = (size_t)b * HWSZ + sp;
    if (n == 0) {
        dep[gpix] = v > 0.f ? v : 0.f;                    // cspn_depth (ReLU)
    } else if (n < 9) {
        graw[gpix * 8 + (n - 1)] = v;                     // g (BN only)
    } else if (n < 40) {
        mid[((size_t)b * 32 + (n - 8)) * HWSZ + sp] = v > 0.f ? v : 0.f;  // x3
    }
}

__global__ void __launch_bounds__(128)
k_conv2(const __bf16* __restrict__ x1, const __bf16* __restrict__ w2,
        const float* __restrict__ bias2, float* __restrict__ dep,
        float* __restrict__ graw, float* __restrict__ mid) {
    __shared__ unsigned wlds[6912];                      // 9*3*32*16 bf16
    const unsigned* wsrc = (const unsigned*)w2;
    for (int i = threadIdx.x; i < 6912; i += 128) wlds[i] = wsrc[i];
    __syncthreads();

    const int wave = threadIdx.x >> 5;
    const int lane = threadIdx.x & 31;
    const int half = lane >> 4;
    const int m    = lane & 15;
    const int b  = blockIdx.z;
    const int y  = blockIdx.y;
    const int xw = blockIdx.x * 64 + wave * 16;

    v8f a0 = {}, a1 = {}, a2 = {};
    const uint4* wq  = (const uint4*)wlds;
    const uint4* ap0 = (const uint4*)(x1 + ((size_t)((b * HH + y) * WW + xw + m)) * CC);

    const bool interior = (y >= 1) && (y <= HH - 2) && (xw >= 1) && (xw <= WW - 17);
    if (interior) {
#pragma unroll
        for (int t = 0; t < 9; ++t) {
            const int dy = t / 3 - 1, dx = t % 3 - 1;
            const int off = (dy * WW + dx) * 4;          // uint4 units
            union { v16bf v; uint4 q[2]; } af;
            af.q[0] = ap0[off + half];
            af.q[1] = ap0[off + 2 + half];
            union { v16bf v; uint4 q[2]; } bf0, bf1, bf2;
            const int i0 = ((t * 3 + 0) * 32 + lane) * 2;
            const int i1 = ((t * 3 + 1) * 32 + lane) * 2;
            const int i2 = ((t * 3 + 2) * 32 + lane) * 2;
            bf0.q[0] = wq[i0]; bf0.q[1] = wq[i0 + 1];
            bf1.q[0] = wq[i1]; bf1.q[1] = wq[i1 + 1];
            bf2.q[0] = wq[i2]; bf2.q[1] = wq[i2 + 1];
            a0 = __builtin_amdgcn_wmma_f32_16x16x32_bf16(false, af.v, false, bf0.v,
                                                         (short)0, a0, false, false);
            a1 = __builtin_amdgcn_wmma_f32_16x16x32_bf16(false, af.v, false, bf1.v,
                                                         (short)0, a1, false, false);
            a2 = __builtin_amdgcn_wmma_f32_16x16x32_bf16(false, af.v, false, bf2.v,
                                                         (short)0, a2, false, false);
        }
    } else {
        for (int t = 0; t < 9; ++t) {
            int dy = t / 3 - 1, dx = t % 3 - 1;
            int py = y + dy, px = xw + m + dx;
            union { v16bf v; uint4 q[2]; } af;
            if (py >= 0 && py < HH && px >= 0 && px < WW) {
                const uint4* ap = (const uint4*)(x1 + ((size_t)((b * HH + py) * WW + px)) * CC);
                af.q[0] = ap[half];
                af.q[1] = ap[2 + half];
            } else {
                af.q[0] = make_uint4(0, 0, 0, 0);
                af.q[1] = make_uint4(0, 0, 0, 0);
            }
            union { v16bf v; uint4 q[2]; } bf0, bf1, bf2;
            int i0 = ((t * 3 + 0) * 32 + lane) * 2;
            int i1 = ((t * 3 + 1) * 32 + lane) * 2;
            int i2 = ((t * 3 + 2) * 32 + lane) * 2;
            bf0.q[0] = wq[i0]; bf0.q[1] = wq[i0 + 1];
            bf1.q[0] = wq[i1]; bf1.q[1] = wq[i1 + 1];
            bf2.q[0] = wq[i2]; bf2.q[1] = wq[i2 + 1];
            a0 = __builtin_amdgcn_wmma_f32_16x16x32_bf16(false, af.v, false, bf0.v,
                                                         (short)0, a0, false, false);
            a1 = __builtin_amdgcn_wmma_f32_16x16x32_bf16(false, af.v, false, bf1.v,
                                                         (short)0, a1, false, false);
            a2 = __builtin_amdgcn_wmma_f32_16x16x32_bf16(false, af.v, false, bf2.v,
                                                         (short)0, a2, false, false);
        }
    }
    const float c0 = bias2[m], c1 = bias2[m + 16], c2 = bias2[m + 32];
#pragma unroll
    for (int r = 0; r < 8; ++r) {
        int mm = r + 8 * half;
        int sp = y * WW + xw + mm;
        route(m,      b, sp, a0[r] + c0, dep, graw, mid);
        route(m + 16, b, sp, a1[r] + c1, dep, graw, mid);
        route(m + 32, b, sp, a2[r] + c2, dep, graw, mid);
    }
}

// ---------------------------------------------------------------------------
// K3: g-normalize + build 9-channel guide [g0..g3, 1-sum, g4..g7], NHWC9
// ---------------------------------------------------------------------------
__global__ void k_norm(const float* __restrict__ graw, float* __restrict__ gk9, int total) {
    int i = blockIdx.x * blockDim.x + threadIdx.x;
    if (i >= total) return;
    float g[8]; float sa = 0.f, ss = 0.f;
    const float* p = graw + (size_t)i * 8;
#pragma unroll
    for (int j = 0; j < 8; ++j) { g[j] = p[j]; sa += fabsf(g[j]); }
    float inv = 1.f / sa;
#pragma unroll
    for (int j = 0; j < 8; ++j) { g[j] *= inv; ss += g[j]; }
    float gm = 1.f - ss;
    float* o = gk9 + (size_t)i * 9;
    o[0] = g[0]; o[1] = g[1]; o[2] = g[2]; o[3] = g[3];
    o[4] = gm;
    o[5] = g[4]; o[6] = g[5]; o[7] = g[6]; o[8] = g[7];
}

// ---------------------------------------------------------------------------
// K4: encoder conv (9->9, 3x3) + CSPN step -> guided depth into mid channel 0
// ---------------------------------------------------------------------------
__global__ void __launch_bounds__(256)
k_cspn(const float* __restrict__ gk9, const float* __restrict__ dep,
       const float* __restrict__ enc, float* __restrict__ mid) {
    __shared__ float e[729];
    for (int i = threadIdx.x; i < 729; i += 256) e[i] = enc[i];
    __syncthreads();
    int idx = blockIdx.x * 256 + threadIdx.x;
    if (idx >= BB * HWSZ) return;
    int b = idx / HWSZ, sp = idx % HWSZ;
    int y = sp / WW, x = sp % WW;

    float nb[81]; float dpatch[9];
#pragma unroll
    for (int ty = 0; ty < 3; ++ty)
#pragma unroll
        for (int tx = 0; tx < 3; ++tx) {
            int tt = ty * 3 + tx;
            int yy = y + ty - 1, xx = x + tx - 1;
            bool ok = (yy >= 0 && yy < HH && xx >= 0 && xx < WW);
            size_t gp = (size_t)b * HWSZ + yy * WW + xx;
            dpatch[tt] = ok ? dep[gp] : 0.f;
#pragma unroll
            for (int i = 0; i < 9; ++i) nb[tt * 9 + i] = ok ? gk9[gp * 9 + i] : 0.f;
        }
    float guided = 0.f;
#pragma unroll
    for (int j = 0; j < 9; ++j) {
        float kj = 0.f;
#pragma unroll
        for (int i = 0; i < 9; ++i)
#pragma unroll
            for (int tt = 0; tt < 9; ++tt)
                kj += e[(j * 9 + i) * 9 + tt] * nb[tt * 9 + i];
        guided += kj * dpatch[j];
    }
    mid[(size_t)b * 32 * HWSZ + sp] = guided;
}

// ---------------------------------------------------------------------------
// K5: 2x bilinear upsample (align-corners), (B,32,256,256) -> (B,32,512,512).
// Output streamed with non-temporal stores (keeps intermediates in L2).
// ---------------------------------------------------------------------------
__global__ void k_up(const float* __restrict__ mid, float* __restrict__ out, long long total) {
    long long i = (long long)blockIdx.x * 256 + threadIdx.x;
    if (i >= total) return;
    const int WO = 512, HO = 512;
    int ox = (int)(i % WO);
    int oy = (int)((i / WO) % HO);
    int cb = (int)(i / ((long long)WO * HO));          // b*32 + c
    const float s = 255.f / 511.f;
    float fy = oy * s, fx = ox * s;
    int y0 = (int)fy, x0 = (int)fx;
    float wy = fy - y0, wx = fx - x0;
    int y1 = y0 + 1; if (y1 > HH - 1) y1 = HH - 1;
    int x1 = x0 + 1; if (x1 > WW - 1) x1 = WW - 1;
    const float* p = mid + (size_t)cb * HWSZ;
    float v00 = p[y0 * WW + x0], v01 = p[y0 * WW + x1];
    float v10 = p[y1 * WW + x0], v11 = p[y1 * WW + x1];
    float top = v00 * (1.f - wx) + v01 * wx;
    float bot = v10 * (1.f - wx) + v11 * wx;
    __builtin_nontemporal_store(top * (1.f - wy) + bot * wy, &out[i]);
}

// ---------------------------------------------------------------------------
extern "C" void kernel_launch(void* const* d_in, const int* in_sizes, int n_in,
                              void* d_out, int out_size, void* d_ws, size_t ws_size,
                              hipStream_t stream) {
    const float* x       = (const float*)d_in[0];
    const float* guide   = (const float*)d_in[1];
    const float* depth   = (const float*)d_in[2];
    const float* conv1_w = (const float*)d_in[3];
    const float* bn1_g   = (const float*)d_in[4];
    const float* bn1_b   = (const float*)d_in[5];
    const float* bn1_m   = (const float*)d_in[6];
    const float* bn1_v   = (const float*)d_in[7];
    const float* conv2_w = (const float*)d_in[8];
    const float* bn2_g   = (const float*)d_in[9];
    const float* bn2_b   = (const float*)d_in[10];
    const float* bn2_m   = (const float*)d_in[11];
    const float* bn2_v   = (const float*)d_in[12];
    const float* conv3_w = (const float*)d_in[13];
    const float* bn3_g   = (const float*)d_in[14];
    const float* bn3_b   = (const float*)d_in[15];
    const float* bn3_m   = (const float*)d_in[16];
    const float* bn3_v   = (const float*)d_in[17];
    const float* gen_w   = (const float*)d_in[18];
    const float* bng_g   = (const float*)d_in[19];
    const float* bng_b   = (const float*)d_in[20];
    const float* bng_m   = (const float*)d_in[21];
    const float* bng_v   = (const float*)d_in[22];
    const float* encoder = (const float*)d_in[23];

    char* ws = (char*)d_ws;
    size_t off = 0;
    auto alloc = [&](size_t bytes) -> void* {
        void* p = ws + off;
        off = (off + bytes + 255) & ~(size_t)255;
        return p;
    };
    __bf16* s16   = (__bf16*)alloc((size_t)BB * HWSZ * CC * 2);   // 32 MB NHWC bf16
    __bf16* x1    = (__bf16*)alloc((size_t)BB * HWSZ * CC * 2);   // 32 MB NHWC bf16
    __bf16* w1    = (__bf16*)alloc(25 * 2 * 32 * 16 * 2);
    float*  bias1 = (float*)alloc(32 * 4);
    __bf16* w2b   = (__bf16*)alloc(9 * 3 * 32 * 16 * 2);
    float*  bias2 = (float*)alloc(48 * 4);
    float*  dep   = (float*)alloc((size_t)BB * HWSZ * 4);
    float*  graw  = (float*)alloc((size_t)BB * HWSZ * 8 * 4);
    float*  gk9   = (float*)alloc((size_t)BB * HWSZ * 9 * 4);
    float*  mid   = (float*)alloc((size_t)BB * 32 * HWSZ * 4);

    dim3 gfuse(HWSZ / 32, BB);
    k_fuse<<<gfuse, 256, 0, stream>>>(x, guide, depth, s16);

    k_pack1<<<(25 * 32 * 32 + 255) / 256, 256, 0, stream>>>(conv1_w, bn1_g, bn1_b, bn1_m, bn1_v,
                                                            w1, bias1);
    k_pack2<<<(9 * 48 * 32 + 255) / 256, 256, 0, stream>>>(conv2_w, bn2_g, bn2_b, bn2_m, bn2_v,
                                                           gen_w, bng_g, bng_b, bng_m, bng_v,
                                                           conv3_w, bn3_g, bn3_b, bn3_m, bn3_v,
                                                           w2b, bias2);

    dim3 gconv(WW / 64, HH, BB);                 // 4 waves/block, 16 px/wave
    k_conv1<<<gconv, 128, 0, stream>>>(s16, w1, bias1, x1);
    k_conv2<<<gconv, 128, 0, stream>>>(x1, w2b, bias2, dep, graw, mid);

    int totP = BB * HWSZ;
    k_norm<<<(totP + 255) / 256, 256, 0, stream>>>(graw, gk9, totP);
    k_cspn<<<(totP + 255) / 256, 256, 0, stream>>>(gk9, dep, encoder, mid);

    long long totO = (long long)BB * 32 * 512 * 512;
    k_up<<<(unsigned)((totO + 255) / 256), 256, 0, stream>>>(mid, (float*)d_out, totO);
}
